// SmallStructureGNN_66383014527589
// MI455X (gfx1250) — matrix operand order repaired and checked by reading, not compiled, not run on hardware
//
#include <hip/hip_runtime.h>
#include <math.h>

#define NN   65536
#define EE   1048576
#define ET   (EE + NN)          // edges incl. self loops
#define FIN  12
#define HID  256
#define NH   8
#define CH   32
#define NL   3
#define NG   64

typedef __attribute__((ext_vector_type(16))) __bf16 v16bf;
typedef __attribute__((ext_vector_type(8)))  float  v8f;

union BfPack { v16bf v; uint4 u[2]; };

__device__ __forceinline__ unsigned short f2bf(float f) {
  unsigned int u = __float_as_uint(f);
  unsigned int r = u + 0x7FFFu + ((u >> 16) & 1u);   // round-to-nearest-even
  return (unsigned short)(r >> 16);
}

__device__ __forceinline__ float gelu_exact(float x) {
  return 0.5f * x * (1.0f + erff(x * 0.70710678118654752f));
}

// ordered-int punning float atomic max (works with -inf init)
__device__ __forceinline__ void atomicMaxF(float* addr, float v) {
  if (v >= 0.0f) atomicMax((int*)addr, __float_as_int(v));
  else           atomicMin((unsigned int*)addr, __float_as_uint(v));
}

// ---------------- embed: h = gelu(x @ W_embed + b) ----------------
__global__ void k_embed(const float* __restrict__ x, const float* __restrict__ W,
                        const float* __restrict__ b, float* __restrict__ h) {
  int i = blockIdx.x * blockDim.x + threadIdx.x;   // N*HID
  int n = i >> 8, o = i & 255;
  float acc = b[o];
#pragma unroll
  for (int k = 0; k < FIN; ++k) acc = fmaf(x[n * FIN + k], W[k * HID + o], acc);
  h[i] = gelu_exact(acc);
}

// ---------------- fp32 -> bf16 feature convert ----------------
__global__ void k_f32_to_bf16(const float* __restrict__ s, unsigned short* __restrict__ d, int n) {
  int i = blockIdx.x * blockDim.x + threadIdx.x;
  if (i < n) d[i] = f2bf(s[i]);
}

// W_conv [L][in][out] f32 -> Wt [L][out][in] bf16 (so B K-dim is contiguous)
__global__ void k_conv_w_t(const float* __restrict__ W, unsigned short* __restrict__ Wt) {
  int i = blockIdx.x * blockDim.x + threadIdx.x;   // NL*HID*HID
  int l = i >> 16, r = i & 65535;
  int o = r >> 8, kin = r & 255;
  Wt[i] = f2bf(W[l * HID * HID + kin * HID + o]);
}

// ---------------- WMMA GEMM: hp[N,256] = hb[N,256] @ W[256,256] ----------------
// one wave per 16x16 output tile; K looped in 32-chunks via v_wmma_f32_16x16x32_bf16
__global__ void k_gemm_wmma(const unsigned short* __restrict__ hb,
                            const unsigned short* __restrict__ wbt,  // [out][in] bf16
                            float* __restrict__ hp) {
  int wave  = threadIdx.x >> 5;
  int lane  = threadIdx.x & 31;
  int tile  = blockIdx.x * 8 + wave;     // 65536 tiles total
  int tileN = tile & 15;                 // HID/16
  int tileM = tile >> 4;                 // N/16
  int m     = tileM * 16 + (lane & 15);
  int ncol  = tileN * 16 + (lane & 15);
  int half  = lane >> 4;                 // lane group 0/1

  v8f c = {};
  for (int kc = 0; kc < HID; kc += 32) {
    // A (16x32 bf16): lanes 0-15 hold K=[kc..kc+7],[kc+16..kc+23]; lanes 16-31 +8
    const uint4* pa0 = (const uint4*)(hb + (size_t)m * HID + kc + half * 8);
    const uint4* pa1 = (const uint4*)(hb + (size_t)m * HID + kc + half * 8 + 16);
    // B (32x16 bf16): lanes 0-15 hold K=[kc..kc+7],[kc+16..kc+23]; lanes 16-31 +8
    const uint4* pb  = (const uint4*)(wbt + (size_t)ncol * HID + kc + half * 8);
    BfPack a, b;
    a.u[0] = pa0[0]; a.u[1] = pa1[0];
    b.u[0] = pb[0];  b.u[1] = pb[2];
    c = __builtin_amdgcn_wmma_f32_16x16x32_bf16(false, a.v, false, b.v,
                                                (short)0, c, false, false);
  }
  // C/D: VGPR r -> row M = r + 8*(lane>=16), col N = lane&15
  int nb = tileN * 16 + (lane & 15);
  int mb = tileM * 16 + half * 8;
#pragma unroll
  for (int r = 0; r < 8; ++r) hp[(size_t)(mb + r) * HID + nb] = c[r];
}

// ---------------- attention coefficients ----------------
__global__ void k_att(const float* __restrict__ hp, const float* __restrict__ as,
                      const float* __restrict__ ad, float* __restrict__ als,
                      float* __restrict__ ald) {
  int i = blockIdx.x * blockDim.x + threadIdx.x;   // N*NH
  int n = i >> 3, hh = i & 7;
  const float* p = hp + (size_t)n * HID + hh * CH;
  float s = 0.f, d = 0.f;
#pragma unroll
  for (int cidx = 0; cidx < CH; ++cidx) {
    float v = p[cidx];
    s = fmaf(v, as[hh * CH + cidx], s);
    d = fmaf(v, ad[hh * CH + cidx], d);
  }
  als[i] = s; ald[i] = d;
}

// ---------------- generic fill ----------------
__global__ void k_fill(float* __restrict__ p, float v, long long n) {
  long long i = (long long)blockIdx.x * blockDim.x + threadIdx.x;
  if (i < n) p[i] = v;
}

// ---------------- edge pass 1: logits + segment max ----------------
__global__ void k_logits(const int* __restrict__ ei, const float* __restrict__ als,
                         const float* __restrict__ ald, float* __restrict__ logit,
                         float* __restrict__ mseg) {
  long long i = (long long)blockIdx.x * blockDim.x + threadIdx.x;  // ET*NH
  if (i >= (long long)ET * NH) return;
  int e = (int)(i >> 3), hh = (int)(i & 7);
  int s, d;
  if (e < EE) { s = ei[e]; d = ei[EE + e]; } else { s = d = e - EE; }
  float v = als[s * NH + hh] + ald[d * NH + hh];
  v = v > 0.f ? v : 0.2f * v;                    // leaky_relu 0.2
  logit[i] = v;
  atomicMaxF(&mseg[d * NH + hh], v);
}

// ---------------- edge pass 2: exp + segment sum (overwrites logit) ----------------
__global__ void k_expsum(const int* __restrict__ ei, float* __restrict__ logit,
                         const float* __restrict__ mseg, float* __restrict__ denom) {
  long long i = (long long)blockIdx.x * blockDim.x + threadIdx.x;  // ET*NH
  if (i >= (long long)ET * NH) return;
  int e = (int)(i >> 3), hh = (int)(i & 7);
  int d = (e < EE) ? ei[EE + e] : (e - EE);
  float ev = expf(logit[i] - mseg[d * NH + hh]);
  logit[i] = ev;
  atomicAdd(&denom[d * NH + hh], ev);
}

// ---------------- edge pass 3: weighted message scatter ----------------
__global__ void k_scatter(const int* __restrict__ ei, const float* __restrict__ ev,
                          const float* __restrict__ denom, const float* __restrict__ hp,
                          float* __restrict__ acc) {
  long long i = (long long)blockIdx.x * blockDim.x + threadIdx.x;  // ET*HID
  if (i >= (long long)ET * HID) return;
  int e = (int)(i >> 8), c = (int)(i & 255), hh = c >> 5;
  int s, d;
  if (e < EE) { s = ei[e]; d = ei[EE + e]; } else { s = d = e - EE; }
  float alpha = ev[(size_t)e * NH + hh] / (denom[d * NH + hh] + 1e-16f);
  atomicAdd(&acc[(size_t)d * HID + c], hp[(size_t)s * HID + c] * alpha);
}

// ---------------- bias + LayerNorm + GELU + residual (wave per node) ----------------
__global__ void k_post(const float* __restrict__ acc, const float* __restrict__ bconv,
                       const float* __restrict__ g, const float* __restrict__ bln,
                       float* __restrict__ h) {
  int wave = threadIdx.x >> 5, lane = threadIdx.x & 31;
  int n = blockIdx.x * (blockDim.x >> 5) + wave;
  float v[8];
  const float* p = acc + (size_t)n * HID + lane * 8;
  float s = 0.f;
#pragma unroll
  for (int r = 0; r < 8; ++r) { v[r] = p[r] + bconv[lane * 8 + r]; s += v[r]; }
#pragma unroll
  for (int o = 16; o >= 1; o >>= 1) s += __shfl_xor(s, o, 32);
  float mean = s * (1.0f / HID);
  float q = 0.f;
#pragma unroll
  for (int r = 0; r < 8; ++r) { float dd = v[r] - mean; q = fmaf(dd, dd, q); }
#pragma unroll
  for (int o = 16; o >= 1; o >>= 1) q += __shfl_xor(q, o, 32);
  float inv = rsqrtf(q * (1.0f / HID) + 1e-5f);
  float* ph = h + (size_t)n * HID + lane * 8;
#pragma unroll
  for (int r = 0; r < 8; ++r) {
    int c = lane * 8 + r;
    float y = (v[r] - mean) * inv * g[c] + bln[c];
    ph[r] += gelu_exact(y);
  }
}

// ---------------- pooling ----------------
__global__ void k_count(const int* __restrict__ batch, float* __restrict__ cnt) {
  int n = blockIdx.x * blockDim.x + threadIdx.x;
  if (n < NN) atomicAdd(&cnt[batch[n]], 1.0f);
}

__global__ void k_pool(const float* __restrict__ h, const int* __restrict__ batch,
                       float* __restrict__ psum, float* __restrict__ pmax) {
  int i = blockIdx.x * blockDim.x + threadIdx.x;  // N*HID
  int n = i >> 8, c = i & 255;
  int gi = batch[n];
  float v = h[i];
  atomicAdd(&psum[gi * HID + c], v);
  atomicMaxF(&pmax[gi * HID + c], v);
}

__global__ void k_mlp1(const float* __restrict__ psum, const float* __restrict__ pmax,
                       const float* __restrict__ cnt, const float* __restrict__ W1,
                       const float* __restrict__ b1, float* __restrict__ t) {
  int i = blockIdx.x * blockDim.x + threadIdx.x;  // NG*HID
  int gi = i >> 8, j = i & 255;
  float invc = 1.0f / fmaxf(cnt[gi], 1.0f);
  float acc = b1[j];
  for (int k = 0; k < HID; ++k) acc = fmaf(psum[gi * HID + k] * invc, W1[k * HID + j], acc);
  for (int k = 0; k < HID; ++k) acc = fmaf(pmax[gi * HID + k], W1[(HID + k) * HID + j], acc);
  t[i] = gelu_exact(acc);
}

__global__ void k_mlp2(const float* __restrict__ t, const float* __restrict__ W2,
                       const float* __restrict__ b2, float* __restrict__ out) {
  int i = blockIdx.x * blockDim.x + threadIdx.x;  // NG*HID
  int gi = i >> 8, j = i & 255;
  float acc = b2[j];
  for (int k = 0; k < HID; ++k) acc = fmaf(t[gi * HID + k], W2[k * HID + j], acc);
  out[i] = acc;
}

extern "C" void kernel_launch(void* const* d_in, const int* in_sizes, int n_in,
                              void* d_out, int out_size, void* d_ws, size_t ws_size,
                              hipStream_t stream) {
  const float* x     = (const float*)d_in[0];
  const int*   ei    = (const int*)d_in[1];
  const int*   batch = (const int*)d_in[2];
  const float* Wemb  = (const float*)d_in[3];
  const float* bemb  = (const float*)d_in[4];
  const float* Wconv = (const float*)d_in[5];
  const float* attS  = (const float*)d_in[6];
  const float* attD  = (const float*)d_in[7];
  const float* bconv = (const float*)d_in[8];
  const float* lng   = (const float*)d_in[9];
  const float* lnb   = (const float*)d_in[10];
  const float* W1    = (const float*)d_in[11];
  const float* b1    = (const float*)d_in[12];
  const float* W2    = (const float*)d_in[13];
  const float* b2    = (const float*)d_in[14];
  float* out = (float*)d_out;

  char* ws = (char*)d_ws;
  size_t off = 0;
  auto carve = [&](size_t bytes) -> char* {
    char* p = ws + off;
    off = (off + bytes + 255) & ~(size_t)255;
    return p;
  };
  float*          h     = (float*)carve((size_t)NN * HID * 4);
  unsigned short* hb    = (unsigned short*)carve((size_t)NN * HID * 2);
  unsigned short* wbt   = (unsigned short*)carve((size_t)NL * HID * HID * 2);
  float*          hp    = (float*)carve((size_t)NN * HID * 4);
  float*          als   = (float*)carve((size_t)NN * NH * 4);
  float*          ald   = (float*)carve((size_t)NN * NH * 4);
  float*          mseg  = (float*)carve((size_t)NN * NH * 4);
  float*          denom = (float*)carve((size_t)NN * NH * 4);
  float*          logit = (float*)carve((size_t)ET * NH * 4);
  float*          acc   = (float*)carve((size_t)NN * HID * 4);
  float*          psum  = (float*)carve((size_t)NG * HID * 4);
  float*          pmax  = (float*)carve((size_t)NG * HID * 4);
  float*          cnt   = (float*)carve((size_t)NG * 4);
  float*          tbuf  = (float*)carve((size_t)NG * HID * 4);

  const int TPB = 256;
  const long long nNH  = (long long)NN * HID;          // 16.7M
  const long long nEH  = (long long)ET * NH;           // 8.9M
  const long long nEC  = (long long)ET * HID;          // 285M

  // embed + weight pre-transpose/convert
  k_embed<<<(unsigned)(nNH / TPB), TPB, 0, stream>>>(x, Wemb, bemb, h);
  k_conv_w_t<<<(NL * HID * HID) / TPB, TPB, 0, stream>>>(Wconv, wbt);

  for (int l = 0; l < NL; ++l) {
    k_f32_to_bf16<<<(unsigned)(nNH / TPB), TPB, 0, stream>>>(h, hb, (int)nNH);
    // 65536 16x16 tiles, 8 waves/block
    k_gemm_wmma<<<(NN / 16) * (HID / 16) / 8, TPB, 0, stream>>>(hb, wbt + (size_t)l * HID * HID, hp);
    k_att<<<(NN * NH) / TPB, TPB, 0, stream>>>(hp, attS + l * NH * CH, attD + l * NH * CH, als, ald);

    k_fill<<<(unsigned)((NN * NH + TPB - 1) / TPB), TPB, 0, stream>>>(mseg, -INFINITY, (long long)NN * NH);
    k_fill<<<(unsigned)((NN * NH + TPB - 1) / TPB), TPB, 0, stream>>>(denom, 0.0f, (long long)NN * NH);
    k_fill<<<(unsigned)(nNH / TPB), TPB, 0, stream>>>(acc, 0.0f, nNH);

    k_logits<<<(unsigned)((nEH + TPB - 1) / TPB), TPB, 0, stream>>>(ei, als, ald, logit, mseg);
    k_expsum<<<(unsigned)((nEH + TPB - 1) / TPB), TPB, 0, stream>>>(ei, logit, mseg, denom);
    k_scatter<<<(unsigned)((nEC + TPB - 1) / TPB), TPB, 0, stream>>>(ei, logit, denom, hp, acc);

    k_post<<<NN / 8, TPB, 0, stream>>>(acc, bconv + l * HID, lng + l * HID, lnb + l * HID, h);
  }

  // pooling + MLP head
  k_fill<<<(NG * HID + TPB - 1) / TPB, TPB, 0, stream>>>(psum, 0.0f, (long long)NG * HID);
  k_fill<<<(NG * HID + TPB - 1) / TPB, TPB, 0, stream>>>(pmax, -INFINITY, (long long)NG * HID);
  k_fill<<<1, TPB, 0, stream>>>(cnt, 0.0f, (long long)NG);
  k_count<<<NN / TPB, TPB, 0, stream>>>(batch, cnt);
  k_pool<<<(unsigned)(nNH / TPB), TPB, 0, stream>>>(h, batch, psum, pmax);
  k_mlp1<<<(NG * HID) / TPB, TPB, 0, stream>>>(psum, pmax, cnt, W1, b1, tbuf);
  k_mlp2<<<(NG * HID) / TPB, TPB, 0, stream>>>(tbuf, W2, b2, out);
}
